// MultiResTriplane_79164837200480
// MI455X (gfx1250) — compile-verified, matrix operand order checked by reading, not run on hardware
//
#include <hip/hip_runtime.h>

typedef __attribute__((ext_vector_type(16))) _Float16 v16h;
typedef __attribute__((ext_vector_type(8)))  _Float16 v8h;
typedef __attribute__((ext_vector_type(8)))  float    v8f;

#define FD    16
#define WIDTH 32
#define BLOCK 256

// ---------------- prepass: (P,C,H,W) fp32 -> (P,H,W,C) fp16 ----------------
// Makes each bilinear corner 2x global_load_b128 instead of 16 scattered b32.
__global__ void transpose_hwc_f16(const float* __restrict__ in,
                                  _Float16* __restrict__ out,
                                  int nTex, int HW) {
  int t = blockIdx.x * blockDim.x + threadIdx.x;
  if (t >= nTex) return;
  int p   = t / HW;
  int rem = t - p * HW;
  const float* src = in + (size_t)p * FD * HW + rem;
  v8h lo, hi;
#pragma unroll
  for (int c = 0; c < 8; ++c) lo[c] = (_Float16)src[(size_t)c * HW];
#pragma unroll
  for (int c = 0; c < 8; ++c) hi[c] = (_Float16)src[(size_t)(c + 8) * HW];
  v8h* dst = (v8h*)(out + (size_t)t * FD);
  dst[0] = lo;
  dst[1] = hi;
}

__device__ __forceinline__ v8h splat8(_Float16 x) {
  v8h r;
#pragma unroll
  for (int i = 0; i < 8; ++i) r[i] = x;
  return r;
}

// ---------------- bilinear sample (align_corners=True, zeros padding) ------
// Packed-f16 accumulation: 8x v_pk_fma_f16 per corner instead of
// 16x v_cvt + 16x v_fma_f32.
__device__ __forceinline__ void sample_add(v8h& a0, v8h& a1,
                                           const _Float16* __restrict__ plane,
                                           int W, int H, float gx, float gy) {
  float px = (gx + 1.0f) * 0.5f * (float)(W - 1);
  float py = (gy + 1.0f) * 0.5f * (float)(H - 1);
  float fx0 = floorf(px), fy0 = floorf(py);
  float wx = px - fx0,    wy = py - fy0;
  int x0 = (int)fx0, y0 = (int)fy0;
#pragma unroll
  for (int cy = 0; cy < 2; ++cy) {
#pragma unroll
    for (int cx = 0; cx < 2; ++cx) {
      int X = x0 + cx, Y = y0 + cy;
      float w = (cx ? wx : 1.0f - wx) * (cy ? wy : 1.0f - wy);
      bool ok = (X >= 0) && (X < W) && (Y >= 0) && (Y < H);
      w = ok ? w : 0.0f;                       // zeros padding
      int Xc = min(max(X, 0), W - 1);
      int Yc = min(max(Y, 0), H - 1);
      const v8h* q = (const v8h*)(plane + ((size_t)Yc * W + Xc) * FD);
      v8h wv = splat8((_Float16)w);
      a0 += wv * q[0];                         // v_pk_fma_f16 x4
      a1 += wv * q[1];                         // v_pk_fma_f16 x4
    }
  }
}

// Load a 16x16x32 f16 WMMA A/B fragment from an LDS row of 64 K-values.
// ISA 7.12.2 16-bit A layout: lane<16 holds K in {0-7,16-23}, lane>=16 holds
// K in {8-15,24-31}; B mirrors by column. kh = (lane>>4)*8.
__device__ __forceinline__ v16h load_frag(const _Float16* row, int c, int kh) {
  v8h lo = *(const v8h*)(row + c * 32 + kh);
  v8h hi = *(const v8h*)(row + c * 32 + 16 + kh);
  v16h r;
#pragma unroll
  for (int i = 0; i < 8; ++i) { r[i] = lo[i]; r[i + 8] = hi[i]; }
  return r;
}

// ---------------- fused sampler + MLP --------------------------------------
__global__ void __launch_bounds__(BLOCK)
triplane_mlp_kernel(const float* __restrict__ coords,
                    const _Float16* __restrict__ xtyt0, const _Float16* __restrict__ xy0,
                    const _Float16* __restrict__ xtyt1, const _Float16* __restrict__ xy1,
                    const _Float16* __restrict__ xtyt2, const _Float16* __restrict__ xy2,
                    const float* __restrict__ w1, const float* __restrict__ b1,
                    const float* __restrict__ w2, const float* __restrict__ b2,
                    float* __restrict__ out, int N) {
  // Row strides padded (72 halfs / 33 floats) to avoid LDS bank conflicts.
  __shared__ _Float16 s_feats[BLOCK * 72];   // 48 feats + zero pad to K=64
  __shared__ _Float16 s_w1t[WIDTH * 72];     // w1 transposed: [n][k], k padded
  __shared__ float    s_h[BLOCK * 33];       // hidden activations
  __shared__ float    s_b1[WIDTH];
  __shared__ float    s_w2[WIDTH * 2];
  __shared__ float    s_b2[2];

  const int tid = threadIdx.x;

  // Stage MLP params (f16 w1ᵀ, K zero-padded 48->64).
  for (int i = tid; i < WIDTH * 64; i += BLOCK) {
    int n = i >> 6, k = i & 63;
    s_w1t[n * 72 + k] = (k < 48) ? (_Float16)w1[k * WIDTH + n] : (_Float16)0.0f;
  }
  if (tid < WIDTH)     s_b1[tid] = b1[tid];
  if (tid < WIDTH * 2) s_w2[tid] = w2[tid];
  if (tid < 2)         s_b2[tid] = b2[tid];

  // Tail: clamp index so EXEC stays all-ones through the WMMA section.
  const int p  = blockIdx.x * BLOCK + tid;
  const int pc = min(p, N - 1);
  const float cx = coords[(size_t)pc * 3 + 0];
  const float cy = coords[(size_t)pc * 3 + 1];
  const float ct = coords[(size_t)pc * 3 + 2];

  v8h acc[6];
#pragma unroll
  for (int i = 0; i < 6; ++i) acc[i] = splat8((_Float16)0.0f);

  // Level 0: rx=128, rt=16
  sample_add(acc[0], acc[1], xtyt0,                 128,  16, cx, ct);
  sample_add(acc[0], acc[1], xtyt0 + 16 * 128 * FD, 128,  16, cy, ct);
  sample_add(acc[0], acc[1], xy0,                   128, 128, cx, cy);
  // Level 1: rx=256, rt=32
  sample_add(acc[2], acc[3], xtyt1,                 256,  32, cx, ct);
  sample_add(acc[2], acc[3], xtyt1 + 32 * 256 * FD, 256,  32, cy, ct);
  sample_add(acc[2], acc[3], xy1,                   256, 256, cx, cy);
  // Level 2: rx=512, rt=64
  sample_add(acc[4], acc[5], xtyt2,                 512,  64, cx, ct);
  sample_add(acc[4], acc[5], xtyt2 + 64 * 512 * FD, 512,  64, cy, ct);
  sample_add(acc[4], acc[5], xy2,                   512, 512, cx, cy);

  // feats -> LDS (already f16; zero pad K 48..63)
  {
    v8h* fp = (v8h*)(&s_feats[tid * 72]);
#pragma unroll
    for (int j = 0; j < 6; ++j) fp[j] = acc[j];
    v8h z = splat8((_Float16)0.0f);
    fp[6] = z;
    fp[7] = z;
  }
  __syncthreads();

  // Layer 1 via WMMA: per wave, 32 points = 2 M-tiles; N=32 = 2 N-tiles;
  // K=64 = 2 chunks -> 8 x v_wmma_f32_16x16x32_f16, fp32 accumulate.
  {
    const int lane = tid & 31;
    const int wv   = tid >> 5;
    const int l15  = lane & 15;
    const int kh   = (lane >> 4) << 3;

    const v16h b00 = load_frag(&s_w1t[l15 * 72],        0, kh);
    const v16h b01 = load_frag(&s_w1t[l15 * 72],        1, kh);
    const v16h b10 = load_frag(&s_w1t[(16 + l15) * 72], 0, kh);
    const v16h b11 = load_frag(&s_w1t[(16 + l15) * 72], 1, kh);

#pragma unroll
    for (int m = 0; m < 2; ++m) {
      const int pt = wv * 32 + m * 16 + l15;
      const v16h a0 = load_frag(&s_feats[pt * 72], 0, kh);
      const v16h a1 = load_frag(&s_feats[pt * 72], 1, kh);
#pragma unroll
      for (int n = 0; n < 2; ++n) {
        v8f cfrag = {};
        cfrag = __builtin_amdgcn_wmma_f32_16x16x32_f16(false, a0, false, n ? b10 : b00,
                                                       (short)0, cfrag, false, false);
        cfrag = __builtin_amdgcn_wmma_f32_16x16x32_f16(false, a1, false, n ? b11 : b01,
                                                       (short)0, cfrag, false, false);
        const float bias = s_b1[n * 16 + l15];
        const int prow = wv * 32 + m * 16 + kh;   // C layout: M = v + (lane>>4)*8
#pragma unroll
        for (int v = 0; v < 8; ++v) {
          float x = cfrag[v] + bias;
          x = (x > 0.0f) ? x : 0.01f * x;          // leaky_relu(0.01)
          s_h[(prow + v) * 33 + n * 16 + l15] = x;
        }
      }
    }
  }
  __syncthreads();

  // Layer 2 (32->2) + sigmoid, scalar per point.
  float o0 = s_b2[0], o1 = s_b2[1];
  const float* hp = &s_h[tid * 33];
#pragma unroll
  for (int j = 0; j < WIDTH; ++j) {
    float hv = hp[j];
    o0 += hv * s_w2[2 * j];
    o1 += hv * s_w2[2 * j + 1];
  }
  o0 = 1.0f / (1.0f + __expf(-o0));
  o1 = 1.0f / (1.0f + __expf(-o1));
  if (p < N) {
    out[2 * (size_t)p]     = o0;
    out[2 * (size_t)p + 1] = o1;
  }
}

extern "C" void kernel_launch(void* const* d_in, const int* in_sizes, int n_in,
                              void* d_out, int out_size, void* d_ws, size_t ws_size,
                              hipStream_t stream) {
  (void)n_in; (void)out_size; (void)ws_size;
  const float* coords = (const float*)d_in[0];
  const float* xtyt0  = (const float*)d_in[1];
  const float* xy0    = (const float*)d_in[2];
  const float* xtyt1  = (const float*)d_in[3];
  const float* xy1    = (const float*)d_in[4];
  const float* xtyt2  = (const float*)d_in[5];
  const float* xy2    = (const float*)d_in[6];
  const float* w1     = (const float*)d_in[7];
  const float* b1     = (const float*)d_in[8];
  const float* w2     = (const float*)d_in[9];
  const float* b2     = (const float*)d_in[10];
  const int N = in_sizes[0] / 3;

  // Channel-last fp16 plane pyramid in workspace: 6,881,280 halfs = 13.8 MB.
  _Float16* ws      = (_Float16*)d_ws;
  _Float16* t_xtyt0 = ws;
  _Float16* t_xy0   = t_xtyt0 + 2 * 16 * 16 * 128;   //   65536
  _Float16* t_xtyt1 = t_xy0   + 1 * 16 * 128 * 128;  //  262144
  _Float16* t_xy1   = t_xtyt1 + 2 * 16 * 32 * 256;   //  262144
  _Float16* t_xtyt2 = t_xy1   + 1 * 16 * 256 * 256;  // 1048576
  _Float16* t_xy2   = t_xtyt2 + 2 * 16 * 64 * 512;   // 1048576

  auto g = [](int n) { return (n + 255) / 256; };
  transpose_hwc_f16<<<g(4096),   256, 0, stream>>>(xtyt0, t_xtyt0, 4096,   16 * 128);
  transpose_hwc_f16<<<g(16384),  256, 0, stream>>>(xy0,   t_xy0,   16384,  128 * 128);
  transpose_hwc_f16<<<g(16384),  256, 0, stream>>>(xtyt1, t_xtyt1, 16384,  32 * 256);
  transpose_hwc_f16<<<g(65536),  256, 0, stream>>>(xy1,   t_xy1,   65536,  256 * 256);
  transpose_hwc_f16<<<g(65536),  256, 0, stream>>>(xtyt2, t_xtyt2, 65536,  64 * 512);
  transpose_hwc_f16<<<g(262144), 256, 0, stream>>>(xy2,   t_xy2,   262144, 512 * 512);

  triplane_mlp_kernel<<<(N + BLOCK - 1) / BLOCK, BLOCK, 0, stream>>>(
      coords, t_xtyt0, t_xy0, t_xtyt1, t_xy1, t_xtyt2, t_xy2,
      w1, b1, w2, b2, (float*)d_out, N);
}
